// NeuralRepModel_40802189312483
// MI455X (gfx1250) — compile-verified, matrix operand order checked by reading, not run on hardware
//
#include <hip/hip_runtime.h>
#include <hip/hip_bf16.h>

// ---------------------------------------------------------------------------
// MI455X (gfx1250) implementation.
//  - Heavy GEMMs: v_wmma_f32_16x16x32_bf16, wave32, 128x128 block tile
//    (8 waves, 32x64 wave tile = 2x4 wmma frags, 8 wmma per 32-K step).
//  - Weights converted once per launch to bf16 pre-TRANSPOSED (Wt[N][K]) so
//    A and B fragments are contiguous 32B GLOBAL loads in one k order shared
//    by A and B, not the ISA table order (lanes 0-15: K=0..15, lanes 16-31: K=16..31).
//    NOTE: fragment addresses are kept as indexed expressions (not pointer
//    arrays) so clang's address-space inference emits global_load_b128, not
//    flat_load_b128 (flat would also tick DScnt and tie up the LDS path).
//  - Output config AND bias presence are compile-time (branch-free epilogue).
//  - WGP-scope global prefetch of the K-strip 2 steps ahead (locality=3).
//  - Mamba scan: 1 block per batch elem, 1 thread per d_inner channel,
//    16-element state in VGPRs, B/C vectors staged in LDS per timestep.
// ---------------------------------------------------------------------------

typedef __attribute__((ext_vector_type(16))) __bf16 v16bf;
typedef __attribute__((ext_vector_type(8)))  float  v8f;
typedef __hip_bfloat16 bf16;

// dims
#define BSZ   1024
#define HID   1024
#define PP    4096
#define SEQ   64
#define DM    64
#define PS    8
#define OL    4096
#define TOK   (BSZ * SEQ)       // 65536 tokens
#define DIN   128               // d_inner
#define NST   16                // d_state
#define DTR   4                 // dt_rank
#define XPROJ_NPAD 64           // 36 -> padded to 64

// ---------------------------------------------------------------------------
// fp32 -> bf16 elementwise convert
__global__ __launch_bounds__(256)
void cvt_bf16_kernel(const float* __restrict__ x, bf16* __restrict__ y, size_t n) {
    size_t i = (size_t)blockIdx.x * 256 + threadIdx.x;
    if (i < n) y[i] = __float2bfloat16(x[i]);
}

// fp32 W[K][N] -> bf16 Wt[Npad][K], zero-padding rows n >= N. 32x32 LDS tile.
__global__ __launch_bounds__(256)
void transpose_bf16_kernel(const float* __restrict__ W, bf16* __restrict__ Wt,
                           int K, int N, int Npad) {
    __shared__ float tile[32][33];
    int kb = blockIdx.x * 32, nb = blockIdx.y * 32;
    int tx = threadIdx.x & 31, ty = threadIdx.x >> 5;   // 32 x 8
    #pragma unroll
    for (int r = 0; r < 32; r += 8) {
        int k = kb + ty + r, n = nb + tx;
        tile[ty + r][tx] = (k < K && n < N) ? W[(size_t)k * N + n] : 0.0f;
    }
    __syncthreads();
    #pragma unroll
    for (int r = 0; r < 32; r += 8) {
        int n = nb + ty + r, k = kb + tx;
        if (n < Npad && k < K) Wt[(size_t)n * K + k] = __float2bfloat16(tile[tx][ty + r]);
    }
}

// ---------------------------------------------------------------------------
// bf16 WMMA GEMM:  C[M][N] = act(A[M][K] @ Bt[N][K]^T + bias)
// Requirements: M%128==0, N%NT==0, K%32==0 (guaranteed by padding).
// 256 threads = 8 waves (4 M x 2 N); block tile 128(M) x NT(N);
// wave tile 32 x NT/2; NT in {64, 128}.
template <int RELU, int NT, bool HASF, bool HASB, bool HASBIAS>
__global__ __launch_bounds__(256)
void gemm_bf16_wmma(const bf16* __restrict__ A, const bf16* __restrict__ Bt,
                    const float* __restrict__ bias,
                    float* __restrict__ Cf, bf16* __restrict__ Cb,
                    int M, int N, int K) {
    constexpr int BJ = NT / 32;            // B fragments per wave (2 or 4)
    const int tid  = threadIdx.x;
    const int wave = tid >> 5;
    const int lane = tid & 31;
    const int bm = blockIdx.y * 128;
    const int bn = blockIdx.x * NT;
    const int wm = bm + (wave & 3) * 32;          // wave's 32 rows
    const int wn = bn + (wave >> 2) * (NT / 2);   // wave's NT/2 cols

    const int lr = lane & 15;              // row-in-tile selector
    const int lk = (lane >> 4) * 16;       // K-half per lane group

    v8f acc[2][BJ];
    #pragma unroll
    for (int i = 0; i < 2; i++)
        #pragma unroll
        for (int j = 0; j < BJ; j++)
            #pragma unroll
            for (int v = 0; v < 8; v++) acc[i][j][v] = 0.0f;

    for (int k0 = 0; k0 < K; k0 += 32) {
        v16bf afrag[2], bfrag[BJ];
        #pragma unroll
        for (int i = 0; i < 2; i++)
            afrag[i] = *(const v16bf*)(A + (size_t)(wm + i * 16 + lr) * K + k0 + lk);
        #pragma unroll
        for (int j = 0; j < BJ; j++)
            bfrag[j] = *(const v16bf*)(Bt + (size_t)(wn + j * 16 + lr) * K + k0 + lk);
        // WGP-scope prefetch 2 k-steps ahead (speculative: OOB tail dropped)
        #pragma unroll
        for (int i = 0; i < 2; i++)
            __builtin_prefetch(A + (size_t)(wm + i * 16 + lr) * K + k0 + lk + 64, 0, 3);
        #pragma unroll
        for (int j = 0; j < BJ; j++)
            __builtin_prefetch(Bt + (size_t)(wn + j * 16 + lr) * K + k0 + lk + 64, 0, 3);
        #pragma unroll
        for (int i = 0; i < 2; i++)
            #pragma unroll
            for (int j = 0; j < BJ; j++)
                acc[i][j] = __builtin_amdgcn_wmma_f32_16x16x32_bf16(
                    false, afrag[i], false, bfrag[j], (short)0, acc[i][j], false, false);
    }

    // Epilogue. C/D layout: lanes 0-15 -> N=lane, M=v; lanes 16-31 -> N=lane-16, M=v+8.
    const int rofs = (lane >> 4) * 8;
    #pragma unroll
    for (int i = 0; i < 2; i++) {
        #pragma unroll
        for (int j = 0; j < BJ; j++) {
            const int col = wn + j * 16 + (lane & 15);
            const int row0 = wm + i * 16 + rofs;
            const float bv = HASBIAS ? bias[col] : 0.0f;
            float* pf = HASF ? (Cf + (size_t)row0 * N + col) : nullptr;
            bf16*  pb = HASB ? (Cb + (size_t)row0 * N + col) : nullptr;
            #pragma unroll
            for (int v = 0; v < 8; v++) {
                float val = acc[i][j][v] + bv;
                if (RELU) val = val > 0.0f ? val : 0.0f;
                if (HASF) { *pf = val; pf += N; }
                if (HASB) { *pb = __float2bfloat16(val); pb += N; }
            }
        }
    }
}

// ---------------------------------------------------------------------------
// Causal depthwise conv1d (D_CONV=4) over L + SiLU.  xz: [TOK][2*DIN], xi = cols 0..127.
__global__ __launch_bounds__(256)
void conv_silu_kernel(const float* __restrict__ xz,
                      const float* __restrict__ conv_w,  // [DIN][4]
                      const float* __restrict__ conv_b,  // [DIN]
                      float* __restrict__ xc_f, bf16* __restrict__ xc_b) {
    size_t idx = (size_t)blockIdx.x * 256 + threadIdx.x;
    if (idx >= (size_t)TOK * DIN) return;
    int d = (int)(idx & (DIN - 1));
    size_t t = idx >> 7;            // token = b*SEQ + l
    int l = (int)(t & (SEQ - 1));
    float acc = conv_b[d];
    #pragma unroll
    for (int k = 0; k < 4; k++) {
        int ls = l + k - 3;
        if (ls >= 0) acc += conv_w[d * 4 + k] * xz[(t + (size_t)(k - 3)) * (2 * DIN) + d];
    }
    float s = acc / (1.0f + __expf(-acc));   // silu
    xc_f[idx] = s;
    xc_b[idx] = __float2bfloat16(s);
}

// ---------------------------------------------------------------------------
// Selective scan: one block per batch element, one thread per d_inner channel.
// dbc: [TOK][64] (cols 0-3 dt, 4-19 B, 20-35 C, 36-63 pad).
__global__ __launch_bounds__(128)
void mamba_scan_kernel(const float* __restrict__ dbc,
                       const float* __restrict__ xc,    // [TOK][DIN]
                       const float* __restrict__ xz,    // [TOK][2*DIN] (z = cols 128..255)
                       const float* __restrict__ dt_w,  // [DTR][DIN]
                       const float* __restrict__ dt_b,  // [DIN]
                       const float* __restrict__ A_log, // [DIN][NST]
                       const float* __restrict__ Dp,    // [DIN]
                       bf16* __restrict__ ybf)          // [TOK][DIN]
{
    int b = blockIdx.x;
    int d = threadIdx.x;
    float Arow[NST], h[NST];
    #pragma unroll
    for (int n = 0; n < NST; n++) { Arow[n] = -__expf(A_log[d * NST + n]); h[n] = 0.0f; }
    float w0 = dt_w[0 * DIN + d], w1 = dt_w[1 * DIN + d];
    float w2 = dt_w[2 * DIN + d], w3 = dt_w[3 * DIN + d];
    float db = dt_b[d], Dd = Dp[d];
    __shared__ float Bs[NST], Cs[NST];

    for (int l = 0; l < SEQ; l++) {
        size_t t = (size_t)b * SEQ + l;
        const float* row = dbc + t * 64;
        if (d < 16)       Bs[d]      = row[4 + d];
        else if (d < 32)  Cs[d - 16] = row[4 + d];   // cols 20..35
        __syncthreads();
        float dtv = row[0] * w0 + row[1] * w1 + row[2] * w2 + row[3] * w3 + db;
        float delta = (dtv > 20.0f) ? dtv : __logf(1.0f + __expf(dtv));  // softplus
        float u = xc[t * DIN + d];
        float y = 0.0f;
        #pragma unroll
        for (int n = 0; n < NST; n++) {
            h[n] = __expf(delta * Arow[n]) * h[n] + delta * Bs[n] * u;
            y += h[n] * Cs[n];
        }
        y += u * Dd;
        float z = xz[t * (2 * DIN) + DIN + d];
        y *= z / (1.0f + __expf(-z));            // * silu(z)
        ybf[t * DIN + d] = __float2bfloat16(y);
        __syncthreads();
    }
}

// ---------------------------------------------------------------------------
// softmax(8) -> 0.99*p + 0.01/8 -> inverse-CDF sample vs u -> one-hot
__global__ __launch_bounds__(256)
void sample_kernel(const float* __restrict__ logits,   // [NR][8] contiguous
                   const float* __restrict__ u,        // [NR]
                   float* __restrict__ out_samples,    // [NR][8]
                   float* __restrict__ out_dist) {     // [NR][8]
    int r = blockIdx.x * 256 + threadIdx.x;
    const int NR = BSZ * PS * SEQ;                     // 524288
    if (r >= NR) return;
    float x[PS], mx = -3.4e38f;
    #pragma unroll
    for (int j = 0; j < PS; j++) { x[j] = logits[(size_t)r * PS + j]; mx = fmaxf(mx, x[j]); }
    float sum = 0.0f;
    #pragma unroll
    for (int j = 0; j < PS; j++) { x[j] = __expf(x[j] - mx); sum += x[j]; }
    float inv = 1.0f / sum;
    float uu = u[r];
    float cdf = 0.0f;
    int idx = 0;
    #pragma unroll
    for (int j = 0; j < PS; j++) {
        float p = 0.99f * x[j] * inv + 0.00125f;
        out_dist[(size_t)r * PS + j] = p;
        cdf += p;
        if (cdf < uu) idx++;
    }
    #pragma unroll
    for (int j = 0; j < PS; j++)
        out_samples[(size_t)r * PS + j] = (j == idx) ? 1.0f : 0.0f;
}

// ---------------------------------------------------------------------------
// Input flattening (setup_inputs() dict insertion order):
// 0 h_state  1 u
// 2 pre.W1  3 pre.b1  4 pre.W2  5 pre.b2  6 pre.W3  7 pre.b3
// 8 m1.W1   9 m1.b1  10 m1.W2  11 m1.b2
// 12.. mamba0: in_proj, conv_w, conv_b, x_proj, dt_w, dt_b, A_log, D, out_proj
// 21.. mamba1: same
extern "C" void kernel_launch(void* const* d_in, const int* in_sizes, int n_in,
                              void* d_out, int out_size, void* d_ws, size_t ws_size,
                              hipStream_t stream) {
    const float* h_state = (const float*)d_in[0];
    const float* u_in    = (const float*)d_in[1];
    const float* preW1 = (const float*)d_in[2];  const float* preB1 = (const float*)d_in[3];
    const float* preW2 = (const float*)d_in[4];  const float* preB2 = (const float*)d_in[5];
    const float* preW3 = (const float*)d_in[6];  const float* preB3 = (const float*)d_in[7];
    const float* m1W1  = (const float*)d_in[8];  const float* m1B1  = (const float*)d_in[9];
    const float* m1W2  = (const float*)d_in[10]; const float* m1B2  = (const float*)d_in[11];
    const float* mam[2][9];
    for (int m = 0; m < 2; m++)
        for (int p = 0; p < 9; p++)
            mam[m][p] = (const float*)d_in[12 + m * 9 + p];
    // mamba param order: 0 in_proj 1 conv_w 2 conv_b 3 x_proj 4 dt_w 5 dt_b 6 A_log 7 D 8 out_proj

    // workspace bump allocator (256B aligned). Total ~350 MB.
    char* ws = (char*)d_ws;
    size_t off = 0;
    auto alloc = [&](size_t bytes) -> void* {
        void* p = ws + off;
        off = (off + bytes + 255) & ~(size_t)255;
        return p;
    };
    bf16* w1t   = (bf16*)alloc((size_t)PP * HID * 2);    // [4096][1024]
    bf16* w2t   = (bf16*)alloc((size_t)PP * PP * 2);
    bf16* w3t   = (bf16*)alloc((size_t)PP * PP * 2);
    bf16* m1w1t = (bf16*)alloc((size_t)OL * PP * 2);
    bf16* m1w2t = (bf16*)alloc((size_t)OL * OL * 2);
    bf16* ipT[2], *xpT[2], *opT[2];
    for (int m = 0; m < 2; m++) {
        ipT[m] = (bf16*)alloc((size_t)(2 * DIN) * DM * 2);     // [256][64]
        xpT[m] = (bf16*)alloc((size_t)XPROJ_NPAD * DIN * 2);   // [64][128]
        opT[m] = (bf16*)alloc((size_t)DM * DIN * 2);           // [64][128]
    }
    bf16*  hbf  = (bf16*)alloc((size_t)BSZ * HID * 2);
    bf16*  xb1  = (bf16*)alloc((size_t)BSZ * PP * 2);
    bf16*  xb2  = (bf16*)alloc((size_t)BSZ * PP * 2);
    bf16*  xb3  = (bf16*)alloc((size_t)BSZ * PP * 2);          // mamba0 input, [TOK][64]
    float* xzf  = (float*)alloc((size_t)TOK * 2 * DIN * 4);
    float* xcf  = (float*)alloc((size_t)TOK * DIN * 4);
    bf16*  xcb  = (bf16*)alloc((size_t)TOK * DIN * 2);
    float* dbcf = (float*)alloc((size_t)TOK * XPROJ_NPAD * 4);
    bf16*  yb   = (bf16*)alloc((size_t)TOK * DIN * 2);
    bf16*  xm0  = (bf16*)alloc((size_t)TOK * DM * 2);
    bf16*  xm1  = (bf16*)alloc((size_t)TOK * DM * 2);
    bf16*  xb4  = (bf16*)alloc((size_t)BSZ * OL * 2);
    float* xf5  = (float*)alloc((size_t)BSZ * OL * 4);
    (void)ws_size; (void)in_sizes; (void)n_in; (void)out_size;

    dim3 t256(256);
    auto tgrid = [](int K, int Npad) { return dim3((K + 31) / 32, (Npad + 31) / 32); };

    // ---- weight prep: fp32 [K][N] -> bf16 [Npad][K] ----
    transpose_bf16_kernel<<<tgrid(HID, PP), t256, 0, stream>>>(preW1, w1t, HID, PP, PP);
    transpose_bf16_kernel<<<tgrid(PP, PP),  t256, 0, stream>>>(preW2, w2t, PP, PP, PP);
    transpose_bf16_kernel<<<tgrid(PP, PP),  t256, 0, stream>>>(preW3, w3t, PP, PP, PP);
    transpose_bf16_kernel<<<tgrid(PP, OL),  t256, 0, stream>>>(m1W1, m1w1t, PP, OL, OL);
    transpose_bf16_kernel<<<tgrid(OL, OL),  t256, 0, stream>>>(m1W2, m1w2t, OL, OL, OL);
    for (int m = 0; m < 2; m++) {
        transpose_bf16_kernel<<<tgrid(DM, 2 * DIN), t256, 0, stream>>>(mam[m][0], ipT[m], DM, 2 * DIN, 2 * DIN);
        transpose_bf16_kernel<<<tgrid(DIN, XPROJ_NPAD), t256, 0, stream>>>(mam[m][3], xpT[m], DIN, DTR + 2 * NST, XPROJ_NPAD);
        transpose_bf16_kernel<<<tgrid(DIN, DM), t256, 0, stream>>>(mam[m][8], opT[m], DIN, DM, DM);
    }
    cvt_bf16_kernel<<<(BSZ * HID + 255) / 256, t256, 0, stream>>>(h_state, hbf, (size_t)BSZ * HID);

    auto ggrid = [](int M, int N, int NT) { return dim3(N / NT, M / 128); };

    // ---- mlp_pre ----
    gemm_bf16_wmma<1, 128, false, true, true><<<ggrid(BSZ, PP, 128), t256, 0, stream>>>(
        hbf, w1t, preB1, nullptr, xb1, BSZ, PP, HID);
    gemm_bf16_wmma<1, 128, false, true, true><<<ggrid(BSZ, PP, 128), t256, 0, stream>>>(
        xb1, w2t, preB2, nullptr, xb2, BSZ, PP, PP);
    gemm_bf16_wmma<0, 128, false, true, true><<<ggrid(BSZ, PP, 128), t256, 0, stream>>>(
        xb2, w3t, preB3, nullptr, xb3, BSZ, PP, PP);

    // ---- two mamba blocks ----
    bf16* min_[2] = { xb3, xm0 };
    bf16* mout[2] = { xm0, xm1 };
    for (int m = 0; m < 2; m++) {
        gemm_bf16_wmma<0, 128, true, false, false><<<ggrid(TOK, 2 * DIN, 128), t256, 0, stream>>>(
            min_[m], ipT[m], nullptr, xzf, nullptr, TOK, 2 * DIN, DM);
        conv_silu_kernel<<<(TOK * DIN + 255) / 256, t256, 0, stream>>>(
            xzf, mam[m][1], mam[m][2], xcf, xcb);
        gemm_bf16_wmma<0, 64, true, false, false><<<ggrid(TOK, XPROJ_NPAD, 64), t256, 0, stream>>>(
            xcb, xpT[m], nullptr, dbcf, nullptr, TOK, XPROJ_NPAD, DIN);
        mamba_scan_kernel<<<BSZ, 128, 0, stream>>>(
            dbcf, xcf, xzf, mam[m][4], mam[m][5], mam[m][6], mam[m][7], yb);
        gemm_bf16_wmma<0, 64, false, true, false><<<ggrid(TOK, DM, 64), t256, 0, stream>>>(
            yb, opT[m], nullptr, nullptr, mout[m], TOK, DM, DIN);
    }

    // ---- mlp1 ----
    gemm_bf16_wmma<1, 128, false, true, true><<<ggrid(BSZ, OL, 128), t256, 0, stream>>>(
        xm1, m1w1t, m1B1, nullptr, xb4, BSZ, OL, PP);
    gemm_bf16_wmma<0, 128, true, false, true><<<ggrid(BSZ, OL, 128), t256, 0, stream>>>(
        xb4, m1w2t, m1B2, xf5, nullptr, BSZ, OL, OL);

    // ---- sample ----
    float* out_samples = (float*)d_out;
    float* out_dist    = out_samples + (size_t)BSZ * OL;
    sample_kernel<<<(BSZ * PS * SEQ + 255) / 256, t256, 0, stream>>>(xf5, u_in, out_samples, out_dist);
}
